// Model_4020089389250
// MI455X (gfx1250) — matrix operand order repaired and checked by reading, not compiled, not run on hardware
//
#include <hip/hip_runtime.h>
#include <math.h>

// ---------------------------------------------------------------------------
// CDNA5 (gfx1250) implementation. All large GEMMs (down/up projections, DFT,
// iDFT, time-MLP, gating-MLP) go through v_wmma_f32_16x16x32_bf16 (wave32).
// ---------------------------------------------------------------------------

typedef __attribute__((ext_vector_type(16))) __bf16 v16bf;
typedef __attribute__((ext_vector_type(8)))  float  v8f;

#define DIV_UP(a, b) (((a) + (b) - 1) / (b))

static __device__ __forceinline__ int imin(int a, int b) { return a < b ? a : b; }

// ------------------------------- WMMA GEMM ---------------------------------
// C[M,N] = act(A[M,K] @ B[K,N] + bias[N]); A,B,C f32 in memory, bf16 in MMA.
// One wave -> 32(M) x 32(N) tile (4 WMMA per K-step); 4 waves/block tile M.
// Batched via blockIdx.z with element strides sA/sB/sC (0 => shared operand).
// Edge policy (branch-free hot loop):
//   - OOB rows/cols: clamp index, accumulate garbage, never store it.
//   - OOB k (tail only): clamped-index load + select-to-zero (v_cndmask).
// Fragment layouts per CDNA5 ISA 7.12.2:
//   A 16x32 bf16 : lane(l) m=l&15, half=l>>4 ; elem e -> k = 16*(e>>3)+8*half+(e&7)
//   B 32x16 bf16 : lane(l) n=l&15, half=l>>4 ; elem e -> k = 16*(e>>3)+8*half+(e&7)
//   C/D 16x16 f32: vgpr r -> m = r + 8*half, n = l&15
__global__ __launch_bounds__(128) void gemm_bf16_wmma(
    const float* __restrict__ A, long long sA,
    const float* __restrict__ Bm, long long sB,
    float* __restrict__ Cm, long long sC,
    const float* __restrict__ bias,
    int M, int N, int K, int act /*0=none,1=erf-gelu*/) {
  const int lane = threadIdx.x & 31;
  const int wave = threadIdx.x >> 5;
  const int half = lane >> 4;
  const int l16  = lane & 15;
  const int mbase = (blockIdx.y * 4 + wave) * 32;
  const int nbase = blockIdx.x * 32;
  if (mbase >= M) return;  // uniform per wave: EXEC stays all-ones for WMMA

  A  += (size_t)blockIdx.z * sA;
  Bm += (size_t)blockIdx.z * sB;
  Cm += (size_t)blockIdx.z * sC;

  const int row0 = imin(mbase + l16, M - 1);
  const int row1 = imin(mbase + 16 + l16, M - 1);
  const int n0 = nbase + l16;
  const int n1 = n0 + 16;

  // clamped base pointers; garbage columns/rows are never stored
  const float* ap0 = A + (size_t)row0 * K + half * 8;
  const float* ap1 = A + (size_t)row1 * K + half * 8;
  const float* bp  = Bm + (size_t)(half << 3) * N;
  const float* bq0 = bp + imin(n0, N - 1);
  const float* bq1 = bp + imin(n1, N - 1);

  v8f acc00 = {}, acc01 = {}, acc10 = {}, acc11 = {};
  const int Kmain = K & ~31;
  int k0 = 0;
  for (; k0 < Kmain; k0 += 32) {
    v16bf a0, a1, b0, b1;
#pragma unroll
    for (int e = 0; e < 8; ++e) {
      a0[e]     = (__bf16)ap0[k0 + e];
      a0[8 + e] = (__bf16)ap0[k0 + 16 + e];
      a1[e]     = (__bf16)ap1[k0 + e];
      a1[8 + e] = (__bf16)ap1[k0 + 16 + e];
    }
#pragma unroll
    for (int e = 0; e < 16; ++e) {
      size_t ko = (size_t)(k0 + ((e >> 3) << 4) + (e & 7)) * N;
      b0[e] = (__bf16)bq0[ko];
      b1[e] = (__bf16)bq1[ko];
    }
    acc00 = __builtin_amdgcn_wmma_f32_16x16x32_bf16(false, a0, false, b0, (short)0, acc00, false, false);
    acc01 = __builtin_amdgcn_wmma_f32_16x16x32_bf16(false, a0, false, b1, (short)0, acc01, false, false);
    acc10 = __builtin_amdgcn_wmma_f32_16x16x32_bf16(false, a1, false, b0, (short)0, acc10, false, false);
    acc11 = __builtin_amdgcn_wmma_f32_16x16x32_bf16(false, a1, false, b1, (short)0, acc11, false, false);
  }
  if (k0 < K) {  // K tail: clamped loads + select-to-zero, still branch-free
    v16bf a0, a1, b0, b1;
#pragma unroll
    for (int e = 0; e < 16; ++e) {
      int k  = k0 + ((e >> 3) << 4) + (half << 3) + (e & 7);
      int kc = imin(k, K - 1);
      float v0 = A[(size_t)row0 * K + kc];
      float v1 = A[(size_t)row1 * K + kc];
      a0[e] = (__bf16)(k < K ? v0 : 0.f);
      a1[e] = (__bf16)(k < K ? v1 : 0.f);
    }
#pragma unroll
    for (int e = 0; e < 16; ++e) {
      int k  = k0 + ((e >> 3) << 4) + (half << 3) + (e & 7);
      int kc = imin(k, K - 1);
      float v0 = Bm[(size_t)kc * N + imin(n0, N - 1)];
      float v1 = Bm[(size_t)kc * N + imin(n1, N - 1)];
      b0[e] = (__bf16)(k < K ? v0 : 0.f);
      b1[e] = (__bf16)(k < K ? v1 : 0.f);
    }
    acc00 = __builtin_amdgcn_wmma_f32_16x16x32_bf16(false, a0, false, b0, (short)0, acc00, false, false);
    acc01 = __builtin_amdgcn_wmma_f32_16x16x32_bf16(false, a0, false, b1, (short)0, acc01, false, false);
    acc10 = __builtin_amdgcn_wmma_f32_16x16x32_bf16(false, a1, false, b0, (short)0, acc10, false, false);
    acc11 = __builtin_amdgcn_wmma_f32_16x16x32_bf16(false, a1, false, b1, (short)0, acc11, false, false);
  }

  const float bia0 = (n0 < N && bias) ? bias[n0] : 0.f;
  const float bia1 = (n1 < N && bias) ? bias[n1] : 0.f;
#pragma unroll
  for (int r = 0; r < 8; ++r) {
    int m0 = mbase + r + half * 8;
    int m1 = m0 + 16;
    if (m0 < M && n0 < N) {
      float v = acc00[r] + bia0;
      if (act) v = 0.5f * v * (1.f + erff(v * 0.70710678118f));
      Cm[(size_t)m0 * N + n0] = v;
    }
    if (m0 < M && n1 < N) {
      float v = acc01[r] + bia1;
      if (act) v = 0.5f * v * (1.f + erff(v * 0.70710678118f));
      Cm[(size_t)m0 * N + n1] = v;
    }
    if (m1 < M && n0 < N) {
      float v = acc10[r] + bia0;
      if (act) v = 0.5f * v * (1.f + erff(v * 0.70710678118f));
      Cm[(size_t)m1 * N + n0] = v;
    }
    if (m1 < M && n1 < N) {
      float v = acc11[r] + bia1;
      if (act) v = 0.5f * v * (1.f + erff(v * 0.70710678118f));
      Cm[(size_t)m1 * N + n1] = v;
    }
  }
}

// ----------------------------- small kernels -------------------------------

__global__ void revin_stats_k(const float* __restrict__ x, float* __restrict__ mean,
                              float* __restrict__ stdv, int Bn, int Ln, int Cn) {
  long long idx = (long long)blockIdx.x * blockDim.x + threadIdx.x;
  if (idx >= (long long)Bn * Cn) return;
  int b = (int)(idx / Cn), c = (int)(idx % Cn);
  const float* p = x + (size_t)b * Ln * Cn + c;
  float s = 0.f, s2 = 0.f;
  for (int t = 0; t < Ln; ++t) { float v = p[(size_t)t * Cn]; s += v; s2 += v * v; }
  float mu = s / Ln;
  float var = fmaxf(s2 / Ln - mu * mu, 0.f);
  mean[idx] = mu;
  stdv[idx] = sqrtf(var + 1e-5f);
}

__global__ void revin_norm_k(const float* __restrict__ x, const float* __restrict__ mean,
                             const float* __restrict__ stdv, const float* __restrict__ rw,
                             const float* __restrict__ rb, float* __restrict__ xn,
                             int Bn, int Ln, int Cn) {
  long long idx = (long long)blockIdx.x * blockDim.x + threadIdx.x;
  long long tot = (long long)Bn * Ln * Cn;
  if (idx >= tot) return;
  int c = (int)(idx % Cn);
  int b = (int)(idx / ((long long)Ln * Cn));
  float mu = mean[(size_t)b * Cn + c], sd = stdv[(size_t)b * Cn + c];
  xn[idx] = (x[idx] - mu) / sd * rw[c] + rb[c];
}

// out[c][r] = in[r][c], batched over blockIdx.z
__global__ void transpose_k(const float* __restrict__ in, float* __restrict__ out,
                            int rows, int cols) {
  __shared__ float tile[32][33];
  size_t base = (size_t)blockIdx.z * rows * cols;
  int r0 = blockIdx.y * 32, c0 = blockIdx.x * 32;
  for (int i = threadIdx.y; i < 32; i += 8) {
    int r = r0 + i, c = c0 + threadIdx.x;
    tile[i][threadIdx.x] = (r < rows && c < cols) ? in[base + (size_t)r * cols + c] : 0.f;
  }
  __syncthreads();
  for (int i = threadIdx.y; i < 32; i += 8) {
    int c = c0 + i, r = r0 + threadIdx.x;
    if (c < cols && r < rows) out[base + (size_t)c * rows + r] = tile[threadIdx.x][i];
  }
}

__global__ void init_dft_k(float* __restrict__ Wre, float* __restrict__ Wim, int F, int Ln) {
  long long idx = (long long)blockIdx.x * blockDim.x + threadIdx.x;
  if (idx >= (long long)F * Ln) return;
  int f = (int)(idx / Ln), t = (int)(idx % Ln);
  int ft = (f * t) & (Ln - 1);  // exact phase reduction (Ln power of 2)
  float ang = 6.283185307179586f * (float)ft / (float)Ln;
  Wre[idx] = cosf(ang);
  Wim[idx] = -sinf(ang);
}

// V[t][f2], f2<F: re-coef = scale*cos ; f2>=F: im-coef = -scale*sin ; scale folds irfft norm
__global__ void init_idft_k(float* __restrict__ V, int Ln, int F) {
  long long idx = (long long)blockIdx.x * blockDim.x + threadIdx.x;
  if (idx >= (long long)Ln * 2 * F) return;
  int f2 = (int)(idx % (2 * F)), t = (int)(idx / (2 * F));
  int f = (f2 < F) ? f2 : f2 - F;
  float scale = ((f == 0) || (f == F - 1)) ? (1.f / Ln) : (2.f / Ln);
  int ft = (f * t) & (Ln - 1);
  float ang = 6.283185307179586f * (float)ft / (float)Ln;
  V[idx] = (f2 < F) ? scale * cosf(ang) : -scale * sinf(ang);
}

__global__ void zero_f32_k(float* p, int n) {
  int i = blockIdx.x * blockDim.x + threadIdx.x;
  if (i < n) p[i] = 0.f;
}

__global__ __launch_bounds__(256) void energy_reduce_k(
    const float* __restrict__ re, const float* __restrict__ im,
    float* __restrict__ energy, int Bn, int F, int Cn) {
  int f = blockIdx.x;
  int nBC = Bn * Cn;
  float s = 0.f;
  for (int i = blockIdx.y * 256 + threadIdx.x; i < nBC; i += gridDim.y * 256) {
    int b = i / Cn, c = i - b * Cn;
    size_t o = ((size_t)b * F + f) * Cn + c;
    float r = re[o], m = im[o];
    s += r * r + m * m;
  }
  __shared__ float sh[256];
  sh[threadIdx.x] = s;
  __syncthreads();
  for (int off = 128; off > 0; off >>= 1) {
    if (threadIdx.x < off) sh[threadIdx.x] += sh[threadIdx.x + off];
    __syncthreads();
  }
  if (threadIdx.x == 0) atomicAdd(&energy[f], sh[0] / (float)nBC);
}

// Single-block: cumulative-energy threshold -> k ; ranks -> binary mask.
__global__ __launch_bounds__(256) void mask_k(const float* __restrict__ energy,
                                              const float* __restrict__ thrp,
                                              float* __restrict__ mask, int F) {
  __shared__ float E[257];
  __shared__ int kshare;
  for (int f = threadIdx.x; f < F; f += 256) E[f] = energy[f];
  __syncthreads();
  if (threadIdx.x == 0) {
    float tot = 0.f;
    for (int f = 0; f < F; ++f) tot += E[f];
    float thr = 0.95f + 0.1f / (1.f + expf(-thrp[0]));
    float cum = 0.f;
    int cnt = 0;  // searchsorted(cum, thr, 'left') == #{cum[i] < thr}
    for (int f = 0; f < F; ++f) { cum += E[f] / tot; if (cum < thr) cnt++; }
    int k = cnt + 1;
    if (k > F) k = F;
    kshare = k;
  }
  __syncthreads();
  int k = kshare;
  for (int f = threadIdx.x; f < F; f += 256) {
    float ef = E[f];
    int rank = 0;
    for (int g = 0; g < F; ++g) {
      float eg = E[g];
      if (eg > ef || (eg == ef && g < f)) rank++;  // stable descending argsort rank
    }
    mask[f] = (rank < k) ? 1.f : 0.f;
  }
}

// sm[b][f2][c] : f2<F -> mask*re ; f2>=F -> mask*im
__global__ void pack_masked_k(const float* __restrict__ re, const float* __restrict__ im,
                              const float* __restrict__ mask, float* __restrict__ sm,
                              int Bn, int F, int Cn) {
  long long idx = (long long)blockIdx.x * blockDim.x + threadIdx.x;
  long long tot = (long long)Bn * 2 * F * Cn;
  if (idx >= tot) return;
  int per = 2 * F * Cn;
  int b = (int)(idx / per);
  int rem = (int)(idx % per);
  int f2 = rem / Cn, c = rem % Cn;
  int f = (f2 < F) ? f2 : f2 - F;
  const float* src = (f2 < F) ? re : im;
  sm[idx] = mask[f] * src[((size_t)b * F + f) * Cn + c];
}

__global__ void ew_mul_inplace_k(float* __restrict__ a, const float* __restrict__ b, long long n) {
  long long i = (long long)blockIdx.x * blockDim.x + threadIdx.x;
  if (i < n) a[i] *= b[i];
}

__global__ void ew_mul2_k(float* __restrict__ d, const float* __restrict__ a,
                          const float* __restrict__ b, long long n) {
  long long i = (long long)blockIdx.x * blockDim.x + threadIdx.x;
  if (i < n) d[i] = a[i] * b[i];
}

// avgpool1d(k=3,s=2,pad=1,count_include_pad) of (in * 0.9^c); in/out [B*C, L]
__global__ void pool_decay_k(const float* __restrict__ in, float* __restrict__ out,
                             int Cn, int Lin, long long total) {
  long long idx = (long long)blockIdx.x * blockDim.x + threadIdx.x;
  if (idx >= total) return;
  int Lout = Lin >> 1;
  int j = (int)(idx % Lout);
  long long bc = idx / Lout;
  int c = (int)(bc % Cn);
  const float* p = in + (size_t)bc * Lin;
  float d = powf(0.9f, (float)c);
  int t = 2 * j;
  float s = p[t];
  if (t - 1 >= 0) s += p[t - 1];
  if (t + 1 < Lin) s += p[t + 1];
  out[idx] = s * d * (1.f / 3.f);
}

__global__ void concat2_k(const float* __restrict__ a, int na, const float* __restrict__ b,
                          int nb, float* __restrict__ d, long long rows) {
  int w = na + nb;
  long long idx = (long long)blockIdx.x * blockDim.x + threadIdx.x;
  if (idx >= rows * w) return;
  long long row = idx / w;
  int j = (int)(idx % w);
  d[idx] = (j < na) ? a[row * na + j] : b[row * nb + (j - na)];
}

__global__ __launch_bounds__(128) void ln_add_jump_k(
    const float* __restrict__ cin, const float* __restrict__ jump,
    const float* __restrict__ g, const float* __restrict__ be,
    float* __restrict__ out, int W) {
  int row = blockIdx.x;
  const float* p = cin + (size_t)row * W;
  float s = 0.f, s2 = 0.f;
  for (int w = threadIdx.x; w < W; w += 128) { float v = p[w]; s += v; s2 += v * v; }
  __shared__ float sh0[128], sh1[128];
  sh0[threadIdx.x] = s; sh1[threadIdx.x] = s2;
  __syncthreads();
  for (int off = 64; off > 0; off >>= 1) {
    if (threadIdx.x < off) { sh0[threadIdx.x] += sh0[threadIdx.x + off]; sh1[threadIdx.x] += sh1[threadIdx.x + off]; }
    __syncthreads();
  }
  float mu = sh0[0] / W;
  float var = fmaxf(sh1[0] / W - mu * mu, 0.f);
  float rinv = rsqrtf(var + 1e-5f);
  for (int w = threadIdx.x; w < W; w += 128)
    out[(size_t)row * W + w] = (p[w] - mu) * rinv * g[w] + be[w] + jump[(size_t)row * W + w];
}

// h[(b*H+hh)][j] = j<C ? trendT[b,c=j,hh] : seasT[b,c=j-C,hh]
__global__ void build_h_k(const float* __restrict__ trendT, const float* __restrict__ seasT,
                          float* __restrict__ h, int Bn, int Hh, int Cn) {
  long long idx = (long long)blockIdx.x * blockDim.x + threadIdx.x;
  long long tot = (long long)Bn * Hh * 2 * Cn;
  if (idx >= tot) return;
  int j = (int)(idx % (2 * Cn));
  long long bh = idx / (2 * Cn);
  int hh = (int)(bh % Hh);
  int b = (int)(bh / Hh);
  int c = (j < Cn) ? j : j - Cn;
  const float* src = (j < Cn) ? trendT : seasT;
  h[idx] = src[((size_t)b * Cn + c) * Hh + hh];
}

// per (b,h): softmax gate over 2 logits, fuse, RevIN denorm.
__global__ __launch_bounds__(128) void final_fuse_k(
    const float* __restrict__ a, const float* __restrict__ glw, const float* __restrict__ glb,
    const float* __restrict__ trendT, const float* __restrict__ seasT,
    const float* __restrict__ mean, const float* __restrict__ stdv,
    const float* __restrict__ rw, const float* __restrict__ rb,
    float* __restrict__ out, int Hh, int Cn) {
  int bh = blockIdx.x;
  int hh = bh % Hh;
  int b = bh / Hh;
  const float* ar = a + (size_t)bh * Cn;
  float s0 = 0.f, s1 = 0.f;
  for (int c = threadIdx.x; c < Cn; c += 128) {
    float v = ar[c];
    s0 += v * glw[2 * c + 0];
    s1 += v * glw[2 * c + 1];
  }
  __shared__ float sh0[128], sh1[128];
  sh0[threadIdx.x] = s0; sh1[threadIdx.x] = s1;
  __syncthreads();
  for (int off = 64; off > 0; off >>= 1) {
    if (threadIdx.x < off) { sh0[threadIdx.x] += sh0[threadIdx.x + off]; sh1[threadIdx.x] += sh1[threadIdx.x + off]; }
    __syncthreads();
  }
  __shared__ float g0s, g1s;
  if (threadIdx.x == 0) {
    float x0 = sh0[0] + glb[0], x1 = sh1[0] + glb[1];
    float m = fmaxf(x0, x1);
    float e0 = expf(x0 - m), e1 = expf(x1 - m);
    float inv = 1.f / (e0 + e1);
    g0s = e0 * inv; g1s = e1 * inv;
  }
  __syncthreads();
  float g0 = g0s, g1 = g1s;
  for (int c = threadIdx.x; c < Cn; c += 128) {
    float t = trendT[((size_t)b * Cn + c) * Hh + hh];
    float s = seasT[((size_t)b * Cn + c) * Hh + hh];
    float z = g0 * t + g1 * s + t + s;
    float o = (z - rb[c]) / (rw[c] + 1e-10f) * stdv[(size_t)b * Cn + c] + mean[(size_t)b * Cn + c];
    out[(size_t)bh * Cn + c] = o;
  }
}

// ------------------------------- host side ---------------------------------

static inline void launch_gemm(hipStream_t s, const float* A, long long sA,
                               const float* Bm, long long sB, float* Cm, long long sC,
                               const float* bias, int M, int N, int K, int act, int batch) {
  dim3 g(DIV_UP(N, 32), DIV_UP(M, 128), batch);
  gemm_bf16_wmma<<<g, 128, 0, s>>>(A, sA, Bm, sB, Cm, sC, bias, M, N, K, act);
}

extern "C" void kernel_launch(void* const* d_in, const int* in_sizes, int n_in,
                              void* d_out, int out_size, void* d_ws, size_t ws_size,
                              hipStream_t stream) {
  (void)in_sizes; (void)n_in; (void)out_size; (void)ws_size;
  constexpr int Bn = 256, Ln = 512, Cn = 321, Hh = 96, F = 257;
  constexpr long long nBC = (long long)Bn * Cn;           // 82176
  constexpr long long nBLC = (long long)Bn * Ln * Cn;     // 42,074,112
  constexpr long long nBFC = (long long)Bn * F * Cn;      // 21,119,232
  constexpr long long nSM = 2 * nBFC;                     // [B, 2F, C]
  constexpr long long BH = (long long)Bn * Hh;            // 24576

  // Inputs: d_in[0] = x_enc; params flattened JAX-style (sorted keys, recursive).
  const float* x_enc   = (const float*)d_in[0];
  const float* down_b0 = (const float*)d_in[1];
  const float* down_b1 = (const float*)d_in[2];
  const float* down_b2 = (const float*)d_in[3];
  const float* down_w0 = (const float*)d_in[4];
  const float* down_w1 = (const float*)d_in[5];
  const float* down_w2 = (const float*)d_in[6];
  const float* gl_b    = (const float*)d_in[7];
  const float* gl_w    = (const float*)d_in[8];
  const float* gm_b1b  = (const float*)d_in[9];
  const float* gm_b1w  = (const float*)d_in[10];
  const float* gm_b2b  = (const float*)d_in[11];
  const float* gm_b2w  = (const float*)d_in[12];
  const float* gm_be   = (const float*)d_in[13];
  const float* gm_cb   = (const float*)d_in[14];
  const float* gm_cw   = (const float*)d_in[15];
  const float* gm_g    = (const float*)d_in[16];
  const float* gm_jb   = (const float*)d_in[17];
  const float* gm_jw   = (const float*)d_in[18];
  const float* revin_b = (const float*)d_in[19];
  const float* revin_w = (const float*)d_in[20];
  const float* thr     = (const float*)d_in[21];
  const float* tm_b1b  = (const float*)d_in[22];
  const float* tm_b1w  = (const float*)d_in[23];
  const float* tm_b2b  = (const float*)d_in[24];
  const float* tm_b2w  = (const float*)d_in[25];
  const float* tm_be   = (const float*)d_in[26];
  const float* tm_cb   = (const float*)d_in[27];
  const float* tm_cw   = (const float*)d_in[28];
  const float* tm_g    = (const float*)d_in[29];
  const float* tm_jb   = (const float*)d_in[30];
  const float* tm_jw   = (const float*)d_in[31];
  const float* up_b0   = (const float*)d_in[32];
  const float* up_b1   = (const float*)d_in[33];
  const float* up_w0   = (const float*)d_in[34];
  const float* up_w1   = (const float*)d_in[35];

  // -------------------- workspace layout (floats), ~1.01 GB ----------------
  float* ws = (float*)d_ws;
  size_t o = 0;
  float* meanb = ws + o; o += nBC;
  float* stdvb = ws + o; o += nBC;
  float* Wre   = ws + o; o += (size_t)F * Ln;
  float* Wim   = ws + o; o += (size_t)F * Ln;
  float* Vc    = ws + o; o += (size_t)Ln * 2 * F;
  float* maskf = ws + o; o += 257;
  float* energ = ws + o; o += 257;
  float* xn    = ws + o; o += nBLC;   // later reused in-place as xr = xn * xr0
  float* xnT   = ws + o; o += nBLC;   // later reused as xrT
  float* re    = ws + o;              // region R3: re+im, later aliased by xr0
  float* im    = re + nBFC; o += nSM;
  float* xr0   = re;                  // alias (re/im dead once sm is packed)
  float* sm    = ws + o; o += nSM;
  float* trendT= ws + o; o += nBC * 96;
  float* seasT = ws + o; o += nBC * 96;
  float* S     = ws + o;              // shared scratch (down/up, then tm, then gm)

  // down/up scratch
  float* enc1 = S;
  float* enc2 = enc1 + nBC * 256;
  float* e0   = enc2 + nBC * 128;
  float* e1   = e0 + nBC * 96;
  float* e2   = e1 + nBC * 48;
  float* cat0 = e2 + nBC * 24;
  float* u0   = cat0 + nBC * 72;
  float* cat1 = u0 + nBC * 48;
  // tm scratch (reuses S after up-path)
  float* tb1   = S;
  float* tb2   = tb1 + nBC * 128;
  float* tc    = tb2 + nBC * 128;
  float* tjump = tc + nBC * 96;
  // gm scratch (reuses S after tm)
  float* hbuf  = S;
  float* gb1   = hbuf + BH * 642;
  float* gb2   = gb1 + BH * 128;
  float* gc    = gb2 + BH * 128;
  float* gjump = gc + BH * 321;
  float* ga    = gjump + BH * 321;

  // ------------------------------ pipeline ---------------------------------
  // RevIN stats + normalize
  revin_stats_k<<<DIV_UP((int)nBC, 256), 256, 0, stream>>>(x_enc, meanb, stdvb, Bn, Ln, Cn);
  revin_norm_k<<<(unsigned)DIV_UP(nBLC, 256), 256, 0, stream>>>(x_enc, meanb, stdvb, revin_w, revin_b, xn, Bn, Ln, Cn);

  // xnT = transpose(xn) : [B][C][L]
  {
    dim3 g(DIV_UP(Cn, 32), DIV_UP(Ln, 32), Bn);
    transpose_k<<<g, dim3(32, 8), 0, stream>>>(xn, xnT, Ln, Cn);
  }

  // DFT / iDFT twiddle tables
  init_dft_k<<<DIV_UP(F * Ln, 256), 256, 0, stream>>>(Wre, Wim, F, Ln);
  init_idft_k<<<DIV_UP(Ln * 2 * F, 256), 256, 0, stream>>>(Vc, Ln, F);

  // ---------------- trend branch: down path + up path ----------------
  launch_gemm(stream, xnT, 0, down_w0, 0, e0, 0, down_b0, (int)nBC, 96, 512, 0, 1);
  pool_decay_k<<<(unsigned)DIV_UP(nBC * 256, 256), 256, 0, stream>>>(xnT, enc1, Cn, 512, nBC * 256);
  launch_gemm(stream, enc1, 0, down_w1, 0, e1, 0, down_b1, (int)nBC, 48, 256, 0, 1);
  pool_decay_k<<<(unsigned)DIV_UP(nBC * 128, 256), 256, 0, stream>>>(enc1, enc2, Cn, 256, nBC * 128);
  launch_gemm(stream, enc2, 0, down_w2, 0, e2, 0, down_b2, (int)nBC, 24, 128, 0, 1);
  concat2_k<<<(unsigned)DIV_UP(nBC * 72, 256), 256, 0, stream>>>(e1, 48, e2, 24, cat0, nBC);
  launch_gemm(stream, cat0, 0, up_w0, 0, u0, 0, up_b0, (int)nBC, 48, 72, 0, 1);
  concat2_k<<<(unsigned)DIV_UP(nBC * 144, 256), 256, 0, stream>>>(e0, 96, u0, 48, cat1, nBC);
  launch_gemm(stream, cat1, 0, up_w1, 0, trendT, 0, up_b1, (int)nBC, 96, 144, 0, 1);

  // ---------------- spectral branch: rFFT -> top-k mask -> irFFT -----------
  // re/im[b,f,c] = sum_t W[f,t] * xn[b,t,c]  (batched over b)
  launch_gemm(stream, Wre, 0, xn, (long long)Ln * Cn, re, (long long)F * Cn, nullptr, F, Cn, Ln, 0, Bn);
  launch_gemm(stream, Wim, 0, xn, (long long)Ln * Cn, im, (long long)F * Cn, nullptr, F, Cn, Ln, 0, Bn);

  zero_f32_k<<<2, 256, 0, stream>>>(energ, 257);
  {
    dim3 g(F, 32);
    energy_reduce_k<<<g, 256, 0, stream>>>(re, im, energ, Bn, F, Cn);
  }
  mask_k<<<1, 256, 0, stream>>>(energ, thr, maskf, F);
  pack_masked_k<<<(unsigned)DIV_UP(nSM, 256), 256, 0, stream>>>(re, im, maskf, sm, Bn, F, Cn);

  // xr0[b,t,c] = sum_f2 Vc[t,f2] * sm[b,f2,c]   (xr0 aliases re/im region)
  launch_gemm(stream, Vc, 0, sm, 2LL * F * Cn, xr0, (long long)Ln * Cn, nullptr, Ln, Cn, 2 * F, 0, Bn);

  // xr = xn * xr0  (in place into xn)
  ew_mul_inplace_k<<<(unsigned)DIV_UP(nBLC, 256), 256, 0, stream>>>(xn, xr0, nBLC);

  // xrT = transpose(xr) -> reuse xnT
  {
    dim3 g(DIV_UP(Cn, 32), DIV_UP(Ln, 32), Bn);
    transpose_k<<<g, dim3(32, 8), 0, stream>>>(xn, xnT, Ln, Cn);
  }

  // ---------------- time MLP (rows = B*C, features L=512) ------------------
  launch_gemm(stream, xnT, 0, tm_b1w, 0, tb1, 0, tm_b1b, (int)nBC, 128, 512, 1, 1);  // gelu
  launch_gemm(stream, xnT, 0, tm_b2w, 0, tb2, 0, tm_b2b, (int)nBC, 128, 512, 0, 1);
  ew_mul2_k<<<(unsigned)DIV_UP(nBC * 128, 256), 256, 0, stream>>>(tb1, tb1, tb2, nBC * 128);
  launch_gemm(stream, tb1, 0, tm_cw, 0, tc, 0, tm_cb, (int)nBC, 96, 128, 1, 1);      // gelu
  launch_gemm(stream, xnT, 0, tm_jw, 0, tjump, 0, tm_jb, (int)nBC, 96, 512, 0, 1);
  ln_add_jump_k<<<(unsigned)nBC, 128, 0, stream>>>(tc, tjump, tm_g, tm_be, seasT, 96);

  // ---------------- gating MLP (rows = B*H, features 2C=642) ---------------
  build_h_k<<<(unsigned)DIV_UP(BH * 642, 256), 256, 0, stream>>>(trendT, seasT, hbuf, Bn, Hh, Cn);
  launch_gemm(stream, hbuf, 0, gm_b1w, 0, gb1, 0, gm_b1b, (int)BH, 128, 642, 1, 1);  // gelu
  launch_gemm(stream, hbuf, 0, gm_b2w, 0, gb2, 0, gm_b2b, (int)BH, 128, 642, 0, 1);
  ew_mul2_k<<<(unsigned)DIV_UP(BH * 128, 256), 256, 0, stream>>>(gb1, gb1, gb2, BH * 128);
  launch_gemm(stream, gb1, 0, gm_cw, 0, gc, 0, gm_cb, (int)BH, 321, 128, 1, 1);      // gelu
  launch_gemm(stream, hbuf, 0, gm_jw, 0, gjump, 0, gm_jb, (int)BH, 321, 642, 0, 1);
  ln_add_jump_k<<<(unsigned)BH, 128, 0, stream>>>(gc, gjump, gm_g, gm_be, ga, 321);

  // ---------------- softmax gate fusion + RevIN denorm ---------------------
  final_fuse_k<<<(unsigned)BH, 128, 0, stream>>>(ga, gl_w, gl_b, trendT, seasT, meanb, stdvb,
                                                 revin_w, revin_b, (float*)d_out, Hh, Cn);
}
